// DecoderMemNN_14929306321428
// MI455X (gfx1250) — matrix operand order repaired and checked by reading, not compiled, not run on hardware
//
#include <hip/hip_runtime.h>
#include <hip/hip_bf16.h>

// ---- problem constants (match reference) ----
#define VV   32000
#define DD   256
#define BB   32
#define MM   1024
#define TT   4
#define HOPS 3

typedef _Float16 v16h __attribute__((ext_vector_type(16)));
typedef _Float16 h8   __attribute__((ext_vector_type(8)));
typedef float    v8f  __attribute__((ext_vector_type(8)));

// ---------------------------------------------------------------------------
// 1) positions: per-batch cumsum of non-pad sentences (tiny, sequential scan)
// ---------------------------------------------------------------------------
__global__ void pos_kernel(const int* __restrict__ story, int* __restrict__ positions) {
    int b = threadIdx.x;
    if (b >= BB) return;
    int cnt = 0;
    for (int m = 0; m < MM; ++m) {
        int tok = story[(m * BB + b) * TT];     // story[m][b][0]
        int nz  = (tok != 0);
        cnt += nz;
        positions[b * MM + m] = nz ? cnt : 0;
    }
}

// ---------------------------------------------------------------------------
// 2) m_story gather: dominant memory traffic (~537 MB of C gathers).
//    64 lanes x float4 per row, 4 rows per 256-thread block.
// ---------------------------------------------------------------------------
__global__ __launch_bounds__(256) void mstory_kernel(
        const int* __restrict__ story, const float* __restrict__ C,
        const float* __restrict__ pos_table, const int* __restrict__ positions,
        float* __restrict__ mstory) {
    int tid = threadIdx.x;
    int r   = blockIdx.x * 4 + (tid >> 6);      // row id over 4*B*M
    int q   = tid & 63;
    int hop = r >> 15;                          // / (B*M)
    int rem = r & 32767;
    int b   = rem >> 10;
    int m   = rem & 1023;
    int d   = q * 4;

    float ax = 0.f, ay = 0.f, az = 0.f, aw = 0.f;
#pragma unroll
    for (int t = 0; t < TT; ++t) {
        int tok = story[(m * BB + b) * TT + t];
        const float4 cv = *(const float4*)(C + ((((size_t)hop * VV + tok)) << 8) + d);
        ax += cv.x; ay += cv.y; az += cv.z; aw += cv.w;
    }
    if (hop == 0) {
        int p = positions[b * MM + m];
        const float4 pv = *(const float4*)(pos_table + (((size_t)p) << 8) + d);
        ax += pv.x; ay += pv.y; az += pv.z; aw += pv.w;
    }
    float4 o; o.x = ax; o.y = ay; o.z = az; o.w = aw;
    *(float4*)(mstory + ((((size_t)hop * BB + b) * MM + m) << 8) + d) = o;
}

// ---------------------------------------------------------------------------
// 3) GRU step: h_new = GRUCell(C0[enc_query[b]], last_hidden[b])
//    One block per batch, thread j owns output feature j (6 dot products).
// ---------------------------------------------------------------------------
__global__ __launch_bounds__(256) void gru_kernel(
        const int* __restrict__ enc_query, const float* __restrict__ C,
        const float* __restrict__ last_hidden,
        const float* __restrict__ w_ih, const float* __restrict__ w_hh,
        const float* __restrict__ b_ih, const float* __restrict__ b_hh,
        float* __restrict__ hidden_out, float* __restrict__ u, float* __restrict__ u0) {
    __shared__ float xs[DD];
    __shared__ float hs[DD];
    int b = blockIdx.x, j = threadIdx.x;
    int qtok = enc_query[b];
    xs[j] = C[(size_t)qtok * DD + j];           // C[0] table
    hs[j] = last_hidden[b * DD + j];
    __syncthreads();

    float gir = b_ih[j], giz = b_ih[DD + j], gin = b_ih[2 * DD + j];
    float ghr = b_hh[j], ghz = b_hh[DD + j], ghn = b_hh[2 * DD + j];
    const float* wi0 = w_ih + (size_t)j * DD;
    const float* wi1 = w_ih + (size_t)(DD + j) * DD;
    const float* wi2 = w_ih + (size_t)(2 * DD + j) * DD;
    const float* wh0 = w_hh + (size_t)j * DD;
    const float* wh1 = w_hh + (size_t)(DD + j) * DD;
    const float* wh2 = w_hh + (size_t)(2 * DD + j) * DD;
    for (int k = 0; k < DD; ++k) {
        float xv = xs[k], hv = hs[k];
        gir += wi0[k] * xv; giz += wi1[k] * xv; gin += wi2[k] * xv;
        ghr += wh0[k] * hv; ghz += wh1[k] * hv; ghn += wh2[k] * hv;
    }
    float rg = 1.0f / (1.0f + expf(-(gir + ghr)));
    float zg = 1.0f / (1.0f + expf(-(giz + ghz)));
    float ng = tanhf(gin + rg * ghn);
    float hn = (1.0f - zg) * ng + zg * hs[j];
    hidden_out[b * DD + j] = hn;
    u[b * DD + j]  = hn;
    u0[b * DD + j] = hn;
}

// ---------------------------------------------------------------------------
// 4) attention dot: prob_lg[b][m] = <m_story[hop][b][m], u[b]>
//    grid (B, 16): 512 blocks streaming m_story at full bandwidth.
//    One wave per 8 rows, 8 floats per lane, shfl reduction (wave32).
// ---------------------------------------------------------------------------
__global__ __launch_bounds__(256) void dot_kernel(
        const float* __restrict__ mh, const float* __restrict__ u,
        float* __restrict__ prob, float* __restrict__ lg_out) {
    __shared__ __align__(16) float us[DD];
    int b = blockIdx.x, tid = threadIdx.x;
    us[tid] = u[b * DD + tid];
    __syncthreads();
    int wv = tid >> 5, lane = tid & 31;
    int mbase = blockIdx.y * 64 + wv * 8;
    const float* rowb = mh + ((size_t)b << 18);          // b*M*D
    int d0 = lane * 8;
    float4 ua = *(const float4*)&us[d0];
    float4 ub = *(const float4*)&us[d0 + 4];
    for (int i = 0; i < 8; ++i) {
        int m = mbase + i;
        const float* row = rowb + ((size_t)m << 8);
        float4 ra = *(const float4*)(row + d0);
        float4 rb = *(const float4*)(row + d0 + 4);
        float p = ra.x * ua.x + ra.y * ua.y + ra.z * ua.z + ra.w * ua.w
                + rb.x * ub.x + rb.y * ub.y + rb.z * ub.z + rb.w * ub.w;
        p += __shfl_down(p, 16, 32);
        p += __shfl_down(p, 8, 32);
        p += __shfl_down(p, 4, 32);
        p += __shfl_down(p, 2, 32);
        p += __shfl_down(p, 1, 32);
        if (lane == 0) {
            prob[b * MM + m] = p;
            if (lg_out) lg_out[b * MM + m] = p;   // hop 2: raw logits output
        }
    }
}

// ---------------------------------------------------------------------------
// 5) softmax over M=1024, in place. One block per batch.
// ---------------------------------------------------------------------------
__global__ __launch_bounds__(256) void softmax_kernel(float* __restrict__ prob) {
    __shared__ float red[256];
    int b = blockIdx.x, tid = threadIdx.x;
    float* pr = prob + b * MM;
    float v0 = pr[tid], v1 = pr[tid + 256], v2 = pr[tid + 512], v3 = pr[tid + 768];
    float mx = fmaxf(fmaxf(v0, v1), fmaxf(v2, v3));
    red[tid] = mx; __syncthreads();
    for (int s = 128; s > 0; s >>= 1) { if (tid < s) red[tid] = fmaxf(red[tid], red[tid + s]); __syncthreads(); }
    mx = red[0]; __syncthreads();
    float e0 = expf(v0 - mx), e1 = expf(v1 - mx), e2 = expf(v2 - mx), e3 = expf(v3 - mx);
    red[tid] = e0 + e1 + e2 + e3; __syncthreads();
    for (int s = 128; s > 0; s >>= 1) { if (tid < s) red[tid] += red[tid + s]; __syncthreads(); }
    float inv = 1.0f / red[0];
    pr[tid] = e0 * inv; pr[tid + 256] = e1 * inv; pr[tid + 512] = e2 * inv; pr[tid + 768] = e3 * inv;
}

// ---------------------------------------------------------------------------
// 6) o_k partials: grid (B, 8); each block handles 128 m's, coalesced over d.
// ---------------------------------------------------------------------------
__global__ __launch_bounds__(256) void okpart_kernel(
        const float* __restrict__ prob, const float* __restrict__ mc,
        float* __restrict__ partial) {
    __shared__ float ps[128];
    int b = blockIdx.x, ch = blockIdx.y, tid = threadIdx.x;
    if (tid < 128) ps[tid] = prob[b * MM + ch * 128 + tid];
    __syncthreads();
    const float* base = mc + (((size_t)b * MM + ch * 128) << 8) + tid;
    float acc = 0.f;
    for (int i = 0; i < 128; ++i) acc += ps[i] * base[(size_t)i << 8];
    partial[(b * 8 + ch) * DD + tid] = acc;
}

// 7) reduce partials; u += o_k; stash o_k of hop 0 for the p_vocab GEMM
__global__ void combine_kernel(const float* __restrict__ partial,
                               float* __restrict__ u, float* __restrict__ ok0) {
    int b = blockIdx.x, d = threadIdx.x;
    float acc = 0.f;
#pragma unroll
    for (int ch = 0; ch < 8; ++ch) acc += partial[(b * 8 + ch) * DD + d];
    if (ok0) ok0[b * DD + d] = acc;
    u[b * DD + d] += acc;
}

// ---------------------------------------------------------------------------
// 8) p_vocab = [concat(u0, ok0)] @ W1_w^T + b  -> [32, 32000] via WMMA f16.
//    Per wave: one 16-wide vocab tile, both 16-row M tiles (B frag reused).
//    A layout (16x32 f16): lane<16 -> K{k0..k0+15},
//                          lane>=16 -> K{k0+16..k0+31}.
//    B layout (32x16):     lane owns column (lane&15), 16 consecutive K from
//                          k0 + 16*(lane>>4) -> one 64B burst per W1 row.
// ---------------------------------------------------------------------------
__global__ __launch_bounds__(256) void gemm_pvocab_kernel(
        const float* __restrict__ u0, const float* __restrict__ ok0,
        const float* __restrict__ W1w, const float* __restrict__ W1b,
        float* __restrict__ out) {
    __shared__ __align__(16) _Float16 Xs[BB][2 * DD];   // 32 x 512 f16 = 32 KB
    const int tid = threadIdx.x;
    for (int idx = tid; idx < BB * 2 * DD; idx += 256) {
        int b = idx >> 9, k = idx & 511;
        float v = (k < DD) ? u0[b * DD + k] : ok0[b * DD + (k - DD)];
        Xs[b][k] = (_Float16)v;
    }
    __syncthreads();

    const int wv   = tid >> 5, lane = tid & 31;
    const int hs   = lane >> 4;                 // half-wave select
    const int nrow = lane & 15;
    const int col0 = blockIdx.x * 128 + wv * 16;
    const int vcol = col0 + nrow;               // vocab column / W1 row

    v8f c0 = {0.f, 0.f, 0.f, 0.f, 0.f, 0.f, 0.f, 0.f};
    v8f c1 = c0;
    const float* __restrict__ wbase = W1w + (size_t)vcol * (2 * DD) + hs * 16;

    for (int k0 = 0; k0 < 2 * DD; k0 += 32) {
        // B fragment: 16 consecutive f32 of this lane's W1 row, cvt -> f16
        const float4* wq = (const float4*)(wbase + k0);
        v16h bf;
#pragma unroll
        for (int i = 0; i < 4; ++i) {
            float4 qv = wq[i];
            bf[4 * i + 0] = (_Float16)qv.x; bf[4 * i + 1] = (_Float16)qv.y;
            bf[4 * i + 2] = (_Float16)qv.z; bf[4 * i + 3] = (_Float16)qv.w;
        }
        // A fragments (two M tiles) from LDS, 16B ds loads
        const int ka = k0 + hs * 16;
        h8 lo0 = *(const h8*)&Xs[nrow][ka];
        h8 hi0 = *(const h8*)&Xs[nrow][ka + 8];
        h8 lo1 = *(const h8*)&Xs[nrow + 16][ka];
        h8 hi1 = *(const h8*)&Xs[nrow + 16][ka + 8];
        v16h a0 = __builtin_shufflevector(lo0, hi0, 0,1,2,3,4,5,6,7,8,9,10,11,12,13,14,15);
        v16h a1 = __builtin_shufflevector(lo1, hi1, 0,1,2,3,4,5,6,7,8,9,10,11,12,13,14,15);

        c0 = __builtin_amdgcn_wmma_f32_16x16x32_f16(false, a0, false, bf, (short)0, c0, false, false);
        c1 = __builtin_amdgcn_wmma_f32_16x16x32_f16(false, a1, false, bf, (short)0, c1, false, false);
    }

    const float bias = W1b[vcol];
    const int mb = hs ? 8 : 0;                   // C/D: VGPR r -> rows r / r+8
#pragma unroll
    for (int r = 0; r < 8; ++r) {
        out[(size_t)(mb + r) * VV + vcol]      = c0[r] + bias;
        out[(size_t)(16 + mb + r) * VV + vcol] = c1[r] + bias;
    }
}

// ---------------------------------------------------------------------------
extern "C" void kernel_launch(void* const* d_in, const int* in_sizes, int n_in,
                              void* d_out, int out_size, void* d_ws, size_t ws_size,
                              hipStream_t stream) {
    const int*   story       = (const int*)d_in[0];
    const int*   enc_query   = (const int*)d_in[1];
    const float* last_hidden = (const float*)d_in[2];
    const float* C           = (const float*)d_in[3];
    const float* pos_table   = (const float*)d_in[4];
    const float* W1_w        = (const float*)d_in[5];
    const float* W1_b        = (const float*)d_in[6];
    const float* gru_w_ih    = (const float*)d_in[7];
    const float* gru_w_hh    = (const float*)d_in[8];
    const float* gru_b_ih    = (const float*)d_in[9];
    const float* gru_b_hh    = (const float*)d_in[10];

    // workspace layout
    char* ws = (char*)d_ws;
    int*   positions = (int*)ws;                               // B*M ints
    float* mstory    = (float*)(ws + 524288);                  // 4*B*M*D f32
    float* u         = (float*)(ws + 524288 + 134217728);      // B*D
    float* u0        = u  + BB * DD;
    float* ok0       = u0 + BB * DD;
    float* prob      = ok0 + BB * DD;                          // B*M
    float* partial   = prob + BB * MM;                         // B*8*D

    // output layout: prob_lg [B,M] | p_vocab [B,V] | hidden [B,D]
    float* lg_out  = (float*)d_out;
    float* pvocab  = lg_out + BB * MM;
    float* hid_out = pvocab + (size_t)BB * VV;

    pos_kernel<<<1, 32, 0, stream>>>(story, positions);
    mstory_kernel<<<(4 * BB * MM) / 4, 256, 0, stream>>>(story, C, pos_table, positions, mstory);
    gru_kernel<<<BB, 256, 0, stream>>>(enc_query, C, last_hidden,
                                       gru_w_ih, gru_w_hh, gru_b_ih, gru_b_hh,
                                       hid_out, u, u0);

    for (int hop = 0; hop < HOPS; ++hop) {
        const float* mh  = mstory + (size_t)hop * BB * MM * DD;
        const float* mh1 = mstory + (size_t)(hop + 1) * BB * MM * DD;
        dot_kernel<<<dim3(BB, 16), 256, 0, stream>>>(mh, u, prob,
                                                     (hop == HOPS - 1) ? lg_out : nullptr);
        softmax_kernel<<<BB, 256, 0, stream>>>(prob);
        okpart_kernel<<<dim3(BB, 8), 256, 0, stream>>>(prob, mh1, partial);
        combine_kernel<<<BB, 256, 0, stream>>>(partial, u, (hop == 0) ? ok0 : nullptr);
    }

    gemm_pvocab_kernel<<<VV / 128, 256, 0, stream>>>(u0, ok0, W1_w, W1_b, pvocab);
}
